// get_model_84035330114060
// MI455X (gfx1250) — matrix operand order repaired and checked by reading, not compiled, not run on hardware
//
#include <hip/hip_runtime.h>

typedef __attribute__((ext_vector_type(16))) _Float16 v16h;
typedef __attribute__((ext_vector_type(8)))  float    v8f;

// ---------------------------------------------------------------------------
// Generic fused GEMM: C = act(((A @ Bt^T) * alpha + bias) * gamma + beta)
//   A : [M,K] row-major (lda), optional batch stride sA (elements)
//   Bt: [N,K] row-major (ldb) -- B is ALWAYS pre-transposed so every fragment
//       load is two contiguous 8-float runs (4x global_load_b128 per k-step)
//   C : [M,N] (ldc); batch stride sC
//   biasM: 0 -> bias indexed by column (N), 1 -> bias indexed by row (M)
// One 16x16 output tile per wave, v_wmma_f32_16x16x32_f16, f32 accumulate.
// OOB rows/cols are handled by address clamping only (their accumulator
// lanes are never stored); only the K-tail needs value selects. All inner
// loops are branch-free so loads batch into clauses.
// ---------------------------------------------------------------------------
__device__ __forceinline__ void load_a_frag(const float* __restrict__ Arow,
                                            int k0, int h, v16h& af)
{
    // per lane: two contiguous 8-float runs: [k0+h*8, +8) and [k0+16+h*8, +8)
#pragma unroll
    for (int j = 0; j < 8; ++j) af[j]     = (_Float16)Arow[k0 + h * 8 + j];
#pragma unroll
    for (int j = 0; j < 8; ++j) af[8 + j] = (_Float16)Arow[k0 + 16 + h * 8 + j];
}

__device__ __forceinline__ void load_b_frag(const float* __restrict__ Brow,
                                            int k0, int h, v16h& bf)
{
    // per lane: two contiguous 8-float runs: [k0+h*8, +8) and [k0+16+h*8, +8)
#pragma unroll
    for (int j = 0; j < 16; ++j) bf[j] = (_Float16)Brow[k0 + (j & 7) + ((j >> 3) * 16) + h * 8];
}

__global__ void wmma_gemm_kernel(const float* __restrict__ A,
                                 const float* __restrict__ Bt,
                                 const float* __restrict__ bias,
                                 const float* __restrict__ gamma,
                                 const float* __restrict__ beta,
                                 float* __restrict__ C,
                                 int M, int N, int K,
                                 int lda, int ldb, int ldc,
                                 long long sA, long long sB, long long sC,
                                 float alpha, int act, int biasM)
{
    const int bz = blockIdx.z;
    A  += (long long)bz * sA;
    Bt += (long long)bz * sB;
    C  += (long long)bz * sC;

    const int lane = threadIdx.x;                      // 0..31 (wave32)
    const int n0 = (blockIdx.x * 4 + threadIdx.y) * 16;
    const int m0 = blockIdx.y * 16;
    if (n0 >= N || m0 >= M) return;                    // wave-uniform exit

    const int h  = lane >> 4;                          // lane-half
    const int ln = lane & 15;

    const int rowA = m0 + ln;
    const int colB = n0 + ln;
    // OOB rows/cols: clamp address only (results discarded at store time).
    const int rowC = rowA < M ? rowA : M - 1;
    const int colC = colB < N ? colB : N - 1;
    const float* Arow = A  + (long long)rowC * lda;
    const float* Brow = Bt + (long long)colC * ldb;

    const int fullK = K & ~31;
    v8f acc = {};

    for (int k0 = 0; k0 < fullK; k0 += 32) {
        v16h af, bf;
        load_a_frag(Arow, k0, h, af);
        load_b_frag(Brow, k0, h, bf);
        acc = __builtin_amdgcn_wmma_f32_16x16x32_f16(false, af, false, bf,
                                                     (short)0, acc, false, false);
    }

    if (fullK < K) {
        // K-tail: unconditional clamped loads + value selects (no branches)
        const int k0 = fullK;
        v16h af, bf;
#pragma unroll
        for (int v = 0; v < 8; ++v) {
            const int kb = k0 + ((v & 3) * 2) + ((v >> 2) * 16) + h * 8;
            const int c0 = kb     < K ? kb     : K - 1;
            const int c1 = kb + 1 < K ? kb + 1 : K - 1;
            const float a0 = Arow[c0];
            const float a1 = Arow[c1];
            af[2 * v]     = (kb     < K) ? (_Float16)a0 : (_Float16)0.f;
            af[2 * v + 1] = (kb + 1 < K) ? (_Float16)a1 : (_Float16)0.f;
        }
#pragma unroll
        for (int j = 0; j < 16; ++j) {
            const int kk = k0 + (j & 7) + ((j >> 3) * 16) + h * 8;
            const int kc = kk < K ? kk : K - 1;
            const float bv = Brow[kc];
            bf[j] = (kk < K) ? (_Float16)bv : (_Float16)0.f;
        }
        acc = __builtin_amdgcn_wmma_f32_16x16x32_f16(false, af, false, bf,
                                                     (short)0, acc, false, false);
    }

    if (colB >= N) return;
    const float bcol = (bias && !biasM) ? bias[colB] : 0.f;
    const float gv   = gamma ? gamma[colB] : 1.f;
    const float tv   = beta  ? beta[colB]  : 0.f;
#pragma unroll
    for (int r = 0; r < 8; ++r) {
        const int m = m0 + r + h * 8;
        if (m < M) {
            float v = acc[r] * alpha + bcol;
            if (bias && biasM) v += bias[m];
            v = v * gv + tv;
            if (act == 1)      v = fmaxf(v, 0.f);
            else if (act == 2) v = 1.f / (1.f + __expf(-v));
            C[(long long)m * ldc + colB] = v;
        }
    }
}

// Wt[n*K + k] = W[k*N + n]
__global__ void transpose_w_kernel(const float* __restrict__ in, float* __restrict__ out,
                                   int K, int N, int total)
{
    int t = blockIdx.x * blockDim.x + threadIdx.x;
    if (t >= total) return;
    int k = t % K; int n = t / K;
    out[t] = in[(long long)k * N + n];
}

// ---------------------------------------------------------------------------
// Farthest point sampling: one block per batch.
// ---------------------------------------------------------------------------
__global__ void fps_kernel(const float* __restrict__ xyz, int N, int S,
                           int* __restrict__ out)
{
    const int b = blockIdx.x;
    const int t = threadIdx.x;  // 256
    __shared__ float dist[2048];
    __shared__ float rv[256];
    __shared__ int   ri[256];
    __shared__ int   far;
    for (int i = t; i < N; i += 256) dist[i] = 1e10f;
    if (t == 0) far = 0;
    __syncthreads();
    for (int j = 0; j < S; ++j) {
        const int f = far;
        if (t == 0) out[b * S + j] = f;
        const float cx = xyz[(b * N + f) * 3 + 0];
        const float cy = xyz[(b * N + f) * 3 + 1];
        const float cz = xyz[(b * N + f) * 3 + 2];
        float best = -1.f; int bi = 0;
        for (int i = t; i < N; i += 256) {
            const float dx = xyz[(b * N + i) * 3 + 0] - cx;
            const float dy = xyz[(b * N + i) * 3 + 1] - cy;
            const float dz = xyz[(b * N + i) * 3 + 2] - cz;
            float d = dx * dx + dy * dy + dz * dz;
            d = fminf(d, dist[i]);
            dist[i] = d;
            if (d > best) { best = d; bi = i; }
        }
        rv[t] = best; ri[t] = bi;
        __syncthreads();
        for (int s = 128; s > 0; s >>= 1) {
            if (t < s) {
                if (rv[t + s] > rv[t] ||
                    (rv[t + s] == rv[t] && ri[t + s] < ri[t])) {
                    rv[t] = rv[t + s]; ri[t] = ri[t + s];
                }
            }
            __syncthreads();
        }
        if (t == 0) far = ri[0];
        __syncthreads();
    }
}

__global__ void transpose_b3n_kernel(const float* __restrict__ in,
                                     float* __restrict__ out, int B, int N, int total)
{
    int t = blockIdx.x * blockDim.x + threadIdx.x;
    if (t >= total) return;
    int c = t % 3; int n = (t / 3) % N; int b = t / (3 * N);
    out[t] = in[(b * 3 + c) * N + n];
}

__global__ void gather_kernel(const float* __restrict__ pts, const int* __restrict__ idx,
                              float* __restrict__ out, int B, int N, int C, int S, int total)
{
    int t = blockIdx.x * blockDim.x + threadIdx.x;
    if (t >= total) return;
    int c = t % C; int s = (t / C) % S; int b = t / (C * S);
    int id = idx[b * S + s];
    out[t] = pts[((long long)b * N + id) * C + c];
}

__global__ void ball_query_kernel(const float* __restrict__ xyz,
                                  const float* __restrict__ nxyz,
                                  int B, int N, int S, int K, float r2,
                                  int* __restrict__ idx, int total)
{
    int t = blockIdx.x * blockDim.x + threadIdx.x;
    if (t >= total) return;
    int q = t % S; int b = t / S;
    const float px = nxyz[(b * S + q) * 3 + 0];
    const float py = nxyz[(b * S + q) * 3 + 1];
    const float pz = nxyz[(b * S + q) * 3 + 2];
    int* ip = idx + ((long long)b * S + q) * K;
    int cnt = 0, first = -1;
    for (int i = 0; i < N; ++i) {
        const float dx = xyz[(b * N + i) * 3 + 0] - px;
        const float dy = xyz[(b * N + i) * 3 + 1] - py;
        const float dz = xyz[(b * N + i) * 3 + 2] - pz;
        const float d = dx * dx + dy * dy + dz * dz;
        if (d <= r2) {
            if (first < 0) first = i;
            ip[cnt++] = i;
            if (cnt >= K) break;
        }
    }
    if (first < 0) first = 0;
    for (int j = cnt; j < K; ++j) ip[j] = first;
}

// g[b, q-q0, k, :] = concat(points[idx], xyz[idx] - new_xyz[q])
__global__ void group_kernel(const float* __restrict__ pts, const float* __restrict__ xyz,
                             const float* __restrict__ nxyz, const int* __restrict__ idx,
                             float* __restrict__ g, int B, int N, int S, int K, int Cp,
                             int q0, int qn, long long total)
{
    long long t = (long long)blockIdx.x * blockDim.x + threadIdx.x;
    if (t >= total) return;
    const int Cg = Cp + 3;
    int c = (int)(t % Cg); long long s = t / Cg;
    int k = (int)(s % K); s /= K;
    int q = (int)(s % qn); int b = (int)(s / qn);
    int qq = q0 + q;
    int id = idx[((long long)b * S + qq) * K + k];
    float v;
    if (c < Cp) v = pts[((long long)b * N + id) * Cp + c];
    else {
        int cc = c - Cp;
        v = xyz[((long long)b * N + id) * 3 + cc] - nxyz[((long long)b * S + qq) * 3 + cc];
    }
    g[t] = v;
}

// out[b, q0+q, c0+c] = max_k in[b, q, k, c]
__global__ void maxpool_kernel(const float* __restrict__ in, float* __restrict__ out,
                               int B, int qn, int K, int C, int S, int Ctot,
                               int q0, int c0, int total)
{
    int t = blockIdx.x * blockDim.x + threadIdx.x;
    if (t >= total) return;
    int c = t % C; int q = (t / C) % qn; int b = t / (C * qn);
    const float* p = in + ((long long)(b * qn + q) * K) * C + c;
    float m = -3.4e38f;
    for (int k = 0; k < K; ++k) m = fmaxf(m, p[(long long)k * C]);
    out[((long long)b * S + q0 + q) * Ctot + c0 + c] = m;
}

__global__ void concat2_kernel(const float* __restrict__ a, const float* __restrict__ bsrc,
                               float* __restrict__ out, int rows, int Ca, int Cb, int total)
{
    int t = blockIdx.x * blockDim.x + threadIdx.x;
    if (t >= total) return;
    const int C = Ca + Cb;
    int c = t % C; int r = t / C;
    out[t] = (c < Ca) ? a[(long long)r * Ca + c] : bsrc[(long long)r * Cb + (c - Ca)];
}

// out[b,n,:] = concat(p1[b,n,:C1], interp3NN(p2)[b,n,:C2]); S==1 -> broadcast p2
__global__ void interp_concat_kernel(const float* __restrict__ xyz1,
                                     const float* __restrict__ xyz2,
                                     const float* __restrict__ p1,
                                     const float* __restrict__ p2,
                                     float* __restrict__ out,
                                     int B, int N1, int S, int C1, int C2, int total)
{
    int t = blockIdx.x * blockDim.x + threadIdx.x;
    if (t >= total) return;
    int n = t % N1; int b = t / N1;
    float* o = out + ((long long)b * N1 + n) * (C1 + C2);
    for (int c = 0; c < C1; ++c) o[c] = p1[((long long)b * N1 + n) * C1 + c];
    if (S == 1) {
        for (int c = 0; c < C2; ++c) o[C1 + c] = p2[(long long)b * C2 + c];
        return;
    }
    const float px = xyz1[(b * N1 + n) * 3 + 0];
    const float py = xyz1[(b * N1 + n) * 3 + 1];
    const float pz = xyz1[(b * N1 + n) * 3 + 2];
    float d0 = 1e30f, d1 = 1e30f, d2 = 1e30f; int i0 = 0, i1 = 0, i2 = 0;
    for (int s = 0; s < S; ++s) {
        const float dx = px - xyz2[(b * S + s) * 3 + 0];
        const float dy = py - xyz2[(b * S + s) * 3 + 1];
        const float dz = pz - xyz2[(b * S + s) * 3 + 2];
        const float d = dx * dx + dy * dy + dz * dz;
        if (d < d0)      { d2 = d1; i2 = i1; d1 = d0; i1 = i0; d0 = d; i0 = s; }
        else if (d < d1) { d2 = d1; i2 = i1; d1 = d; i1 = s; }
        else if (d < d2) { d2 = d; i2 = s; }
    }
    float w0 = 1.f / (d0 + 1e-8f), w1 = 1.f / (d1 + 1e-8f), w2 = 1.f / (d2 + 1e-8f);
    const float ws = w0 + w1 + w2;
    w0 /= ws; w1 /= ws; w2 /= ws;
    const float* P2 = p2 + (long long)b * S * C2;
    for (int c = 0; c < C2; ++c)
        o[C1 + c] = w0 * P2[(long long)i0 * C2 + c] + w1 * P2[(long long)i1 * C2 + c] +
                    w2 * P2[(long long)i2 * C2 + c];
}

__global__ void build_p1_kernel(const float* __restrict__ cls, const float* __restrict__ x,
                                float* __restrict__ out, int N, int total)
{
    int t = blockIdx.x * blockDim.x + threadIdx.x;
    if (t >= total) return;
    int c = t % 22; int n = (t / 22) % N; int b = t / (22 * N);
    float v;
    if (c < 16)      v = cls[b * 16 + c];
    else if (c < 19) v = x[((long long)b * N + n) * 3 + (c - 16)];
    else             v = x[((long long)b * N + n) * 3 + (c - 19)];
    out[t] = v;
}

// x = LayerNorm(x + r) * g + b, d = 128, one block (128 thr) per row
__global__ void ln_res_kernel(float* __restrict__ x, const float* __restrict__ r,
                              const float* __restrict__ g, const float* __restrict__ b)
{
    const int row = blockIdx.x; const int t = threadIdx.x;
    __shared__ float red[128];
    const long long base = (long long)row * 128;
    float v = x[base + t] + r[base + t];
    red[t] = v; __syncthreads();
    for (int s = 64; s > 0; s >>= 1) { if (t < s) red[t] += red[t + s]; __syncthreads(); }
    const float mu = red[0] / 128.f;
    __syncthreads();
    const float dv = v - mu;
    red[t] = dv * dv; __syncthreads();
    for (int s = 64; s > 0; s >>= 1) { if (t < s) red[t] += red[t + s]; __syncthreads(); }
    const float var = red[0] / 128.f;
    x[base + t] = dv * rsqrtf(var + 1e-5f) * g[t] + b[t];
}

// in-place row softmax, one block (256 thr) per row of length L
__global__ void softmax_kernel(float* __restrict__ x, int L)
{
    const int row = blockIdx.x; const int t = threadIdx.x;
    __shared__ float red[256];
    float* X = x + (long long)row * L;
    float mx = -3.4e38f;
    for (int i = t; i < L; i += 256) mx = fmaxf(mx, X[i]);
    red[t] = mx; __syncthreads();
    for (int s = 128; s > 0; s >>= 1) { if (t < s) red[t] = fmaxf(red[t], red[t + s]); __syncthreads(); }
    mx = red[0]; __syncthreads();
    float sum = 0.f;
    for (int i = t; i < L; i += 256) { const float e = __expf(X[i] - mx); X[i] = e; sum += e; }
    red[t] = sum; __syncthreads();
    for (int s = 128; s > 0; s >>= 1) { if (t < s) red[t] += red[t + s]; __syncthreads(); }
    const float inv = 1.f / red[0];
    for (int i = t; i < L; i += 256) X[i] *= inv;
}

__global__ void meanpool_kernel(const float* __restrict__ f, float* __restrict__ out,
                                int B, int N, int C, int total)
{
    int t = blockIdx.x * blockDim.x + threadIdx.x;
    if (t >= total) return;
    int c = t % C; int b = t / C;
    float acc = 0.f;
    for (int n = 0; n < N; ++n) acc += f[((long long)b * N + n) * C + c];
    out[t] = acc / (float)N;
}

// x[(l*B+b)*128+c] = qe[l*128+c]
__global__ void bc_query_kernel(const float* __restrict__ qe, float* __restrict__ x,
                                int B, int total)
{
    int t = blockIdx.x * blockDim.x + threadIdx.x;
    if (t >= total) return;
    int c = t % 128; int i = t / 128; int l = i / B;
    x[t] = qe[l * 128 + c];
}

// y[(b*L+l)*128+c] = x[(l*B+b)*128+c]
__global__ void lb_to_bl_kernel(const float* __restrict__ x, float* __restrict__ y,
                                int L, int B, int total)
{
    int t = blockIdx.x * blockDim.x + threadIdx.x;
    if (t >= total) return;
    int c = t % 128; int i = t / 128; int l = i % L; int b = i / L;
    y[t] = x[((long long)(l * B + b)) * 128 + c];
}

// ---------------------------------------------------------------------------
// Host side
// ---------------------------------------------------------------------------
#define EW(kern, total, ...)                                                   \
    do {                                                                       \
        long long _t = (long long)(total);                                     \
        if (_t > 0) kern<<<(unsigned)((_t + 255) / 256), 256, 0, stream>>>(__VA_ARGS__); \
    } while (0)

static void gemm(hipStream_t stream,
                 const float* A, const float* Bt, const float* bias,
                 const float* gamma, const float* beta, float* C,
                 int M, int N, int K, int lda, int ldb, int ldc,
                 float alpha, int act, int biasM = 0,
                 int nb = 1, long long sA = 0, long long sB = 0, long long sC = 0)
{
    dim3 g((N + 63) / 64, (M + 15) / 16, nb);
    dim3 b(32, 4, 1);
    wmma_gemm_kernel<<<g, b, 0, stream>>>(A, Bt, bias, gamma, beta, C, M, N, K,
                                          lda, ldb, ldc, sA, sB, sC, alpha, act, biasM);
}

// Leaf order: jax pytree flatten (dict keys sorted, lists in order)
enum {
    IN_CLS = 0,
    IN_CONV1 = 1,                 // W,b,beta,gamma
    IN_COORD_W1 = 5, IN_COORD_W2 = 6, IN_COORD_B1 = 7, IN_COORD_B2 = 8,
    IN_COUNT_W1 = 9, IN_COUNT_W2 = 10, IN_COUNT_B1 = 11, IN_COUNT_B2 = 12,
    IN_EXIST_W = 13, IN_EXIST_B = 14,
    IN_FP1 = 15, IN_FP2 = 23, IN_FP3 = 31,
    IN_QUERY = 39,
    IN_SA1 = 40, IN_SA2 = 76, IN_SA3 = 100,
    IN_TR = 112,                  // per layer (16): W1,W2,Wk,Wo,Wq,Wv,b1,b2,bk,bo,bq,bv,ln1b,ln1g,ln2b,ln2g
    IN_XYZ = 144
};

// all weight matrices that need a one-time [K,N] -> [N,K] transpose: {idx, K, N}
static const int WTAB[][3] = {
    {IN_CONV1, 128, 128},
    {IN_COORD_W1, 128, 128}, {IN_COORD_W2, 128, 3},
    {IN_COUNT_W1, 128, 64},  {IN_COUNT_W2, 64, 1},
    {IN_EXIST_W, 128, 1},
    {IN_FP1 + 0, 150, 128},  {IN_FP1 + 4, 128, 128},
    {IN_FP2 + 0, 576, 256},  {IN_FP2 + 4, 256, 128},
    {IN_FP3 + 0, 1536, 256}, {IN_FP3 + 4, 256, 256},
    {IN_SA1 + 0, 6, 32},   {IN_SA1 + 4, 32, 32},  {IN_SA1 + 8, 32, 64},
    {IN_SA1 + 12, 6, 64},  {IN_SA1 + 16, 64, 64}, {IN_SA1 + 20, 64, 128},
    {IN_SA1 + 24, 6, 64},  {IN_SA1 + 28, 64, 96}, {IN_SA1 + 32, 96, 128},
    {IN_SA2 + 0, 323, 128}, {IN_SA2 + 4, 128, 128}, {IN_SA2 + 8, 128, 256},
    {IN_SA2 + 12, 323, 128}, {IN_SA2 + 16, 128, 196}, {IN_SA2 + 20, 196, 256},
    {IN_SA3 + 0, 515, 256}, {IN_SA3 + 4, 256, 512}, {IN_SA3 + 8, 512, 1024},
    {IN_TR + 0, 128, 512}, {IN_TR + 1, 512, 128}, {IN_TR + 2, 128, 128},
    {IN_TR + 3, 128, 128}, {IN_TR + 4, 128, 128}, {IN_TR + 5, 128, 128},
    {IN_TR + 16, 128, 512}, {IN_TR + 17, 512, 128}, {IN_TR + 18, 128, 128},
    {IN_TR + 19, 128, 128}, {IN_TR + 20, 128, 128}, {IN_TR + 21, 128, 128},
};

static void conv_layer(hipStream_t stream, const float* const* in, float* const* wt,
                       int base, const float* A, float* C, int M, int Kc, int Nc)
{
    // reference: relu((A@W + b) * gamma + beta); group order W,b,beta,gamma
    gemm(stream, A, wt[base], in[base + 1], in[base + 3], in[base + 2], C,
         M, Nc, Kc, Kc, Kc, Nc, 1.f, 1);
}

extern "C" void kernel_launch(void* const* d_in, const int* in_sizes, int n_in,
                              void* d_out, int out_size, void* d_ws, size_t ws_size,
                              hipStream_t stream)
{
    (void)in_sizes; (void)n_in; (void)out_size; (void)ws_size;
    const float* const* in = (const float* const*)d_in;
    float* fout = (float*)d_out;
    const int B = 4, N0 = 2048;

    char* w = (char*)d_ws; size_t off = 0;
    auto fbuf = [&](size_t n) -> float* { float* p = (float*)(w + off); off = (off + n * 4 + 255) & ~(size_t)255; return p; };
    auto ibuf = [&](size_t n) -> int*   { int*   p = (int*)(w + off);   off = (off + n * 4 + 255) & ~(size_t)255; return p; };

    float* xb    = fbuf((size_t)B * N0 * 3);
    int*   fidx  = ibuf((size_t)B * 1024);
    float* l1x   = fbuf((size_t)B * 1024 * 3);
    int*   fidx2 = ibuf((size_t)B * 256);
    float* l2x   = fbuf((size_t)B * 256 * 3);
    float* l1p   = fbuf((size_t)B * 1024 * 320);
    float* l2p   = fbuf((size_t)B * 256 * 512);
    float* l3p   = fbuf((size_t)B * 1024);
    float* l2pn  = fbuf((size_t)B * 256 * 256);
    float* l1pn  = fbuf((size_t)B * 1024 * 128);
    int*   idxb  = ibuf((size_t)B * 1024 * 128);
    float* gA    = fbuf((size_t)B * 64 * 128 * 323);   // 10.6M floats
    float* gB    = fbuf((size_t)B * 128 * 128 * 128);  // 8.4M floats
    float* fpb   = fbuf((size_t)B * 1024 * 576);
    float* p1b   = fbuf((size_t)B * N0 * 22);
    float* l0p   = fbuf((size_t)B * N0 * 128);
    float* feat  = fbuf((size_t)B * N0 * 128);
    float* pooled= fbuf((size_t)B * 128);
    float* cnth  = fbuf((size_t)B * 64);
    float* xq    = fbuf((size_t)4000 * 128);
    float* qb    = fbuf((size_t)4000 * 128);
    float* kb    = fbuf((size_t)4000 * 128);
    float* vt    = fbuf((size_t)B * 128 * 1000);       // V^T per batch: [d, m]
    float* ob    = fbuf((size_t)4000 * 128);
    float* tb    = fbuf((size_t)4000 * 128);
    float* hb    = fbuf((size_t)4000 * 512);
    float* sc    = fbuf((size_t)4 * 1000 * 1000);
    float* dbt   = fbuf((size_t)4000 * 128);
    float* hd    = fbuf((size_t)4000 * 128);

    // ---- one-time weight transposes: W[K,N] -> Wt[N,K] ----
    float* wt[160] = {};
    for (size_t e = 0; e < sizeof(WTAB) / sizeof(WTAB[0]); ++e) {
        const int idx = WTAB[e][0], Kc = WTAB[e][1], Nc = WTAB[e][2];
        wt[idx] = fbuf((size_t)Kc * Nc);
        EW(transpose_w_kernel, Kc * Nc, in[idx], wt[idx], Kc, Nc, Kc * Nc);
    }

    // ---- l0: x = transpose(xyz) ----
    EW(transpose_b3n_kernel, B * N0 * 3, in[IN_XYZ], xb, B, N0, B * N0 * 3);

    // ---- SA1 (MSG): npoint=1024 ----
    fps_kernel<<<B, 256, 0, stream>>>(xb, N0, 1024, fidx);
    EW(gather_kernel, B * 1024 * 3, xb, fidx, l1x, B, N0, 3, 1024, B * 1024 * 3);
    {
        const float rr[3] = {0.1f, 0.2f, 0.4f};
        const int   kk[3] = {32, 64, 128};
        const int   ch[3][3] = {{32, 32, 64}, {64, 64, 128}, {64, 96, 128}};
        const int   c0[3] = {0, 64, 192};
        const int S = 1024, CH = 128, Cp = 3;
        for (int br = 0; br < 3; ++br) {
            const int K = kk[br];
            EW(ball_query_kernel, B * S, xb, l1x, B, N0, S, K, rr[br] * rr[br], idxb, B * S);
            for (int q0 = 0; q0 < S; q0 += CH) {
                const int rows = B * CH * K;
                EW(group_kernel, (long long)rows * (Cp + 3), xb, xb, l1x, idxb, gA,
                   B, N0, S, K, Cp, q0, CH, (long long)rows * (Cp + 3));
                const float* cur = gA; float* nxt = gB; int cin = Cp + 3;
                for (int li = 0; li < 3; ++li) {
                    conv_layer(stream, in, wt, IN_SA1 + br * 12 + li * 4, cur, nxt, rows, cin, ch[br][li]);
                    cin = ch[br][li];
                    cur = nxt; nxt = (nxt == gB) ? gA : gB;
                }
                EW(maxpool_kernel, B * CH * cin, gB, l1p, B, CH, K, cin, S, 320, q0, c0[br], B * CH * cin);
            }
        }
    }

    // ---- SA2 (MSG): npoint=256 ----
    fps_kernel<<<B, 256, 0, stream>>>(l1x, 1024, 256, fidx2);
    EW(gather_kernel, B * 256 * 3, l1x, fidx2, l2x, B, 1024, 3, 256, B * 256 * 3);
    {
        const float rr[2] = {0.4f, 0.8f};
        const int   kk[2] = {64, 128};
        const int   ch[2][3] = {{128, 128, 256}, {128, 196, 256}};
        const int   c0[2] = {0, 256};
        const int S = 256, CH = 64, Cp = 320, N1 = 1024;
        for (int br = 0; br < 2; ++br) {
            const int K = kk[br];
            EW(ball_query_kernel, B * S, l1x, l2x, B, N1, S, K, rr[br] * rr[br], idxb, B * S);
            for (int q0 = 0; q0 < S; q0 += CH) {
                const int rows = B * CH * K;
                EW(group_kernel, (long long)rows * (Cp + 3), l1p, l1x, l2x, idxb, gA,
                   B, N1, S, K, Cp, q0, CH, (long long)rows * (Cp + 3));
                const float* cur = gA; float* nxt = gB; int cin = Cp + 3;
                for (int li = 0; li < 3; ++li) {
                    conv_layer(stream, in, wt, IN_SA2 + br * 12 + li * 4, cur, nxt, rows, cin, ch[br][li]);
                    cin = ch[br][li];
                    cur = nxt; nxt = (nxt == gB) ? gA : gB;
                }
                EW(maxpool_kernel, B * CH * cin, gB, l2p, B, CH, K, cin, S, 512, q0, c0[br], B * CH * cin);
            }
        }
    }

    // ---- SA3: global ----
    EW(concat2_kernel, B * 256 * 515, l2x, l2p, gA, B * 256, 3, 512, B * 256 * 515);
    conv_layer(stream, in, wt, IN_SA3 + 0, gA, gB, B * 256, 515, 256);
    conv_layer(stream, in, wt, IN_SA3 + 4, gB, gA, B * 256, 256, 512);
    conv_layer(stream, in, wt, IN_SA3 + 8, gA, gB, B * 256, 512, 1024);
    EW(maxpool_kernel, B * 1024, gB, l3p, B, 1, 256, 1024, 1, 1024, 0, 0, B * 1024);

    // ---- FP3 / FP2 / FP1 ----
    EW(interp_concat_kernel, B * 256, l2x, (const float*)nullptr, l2p, l3p, fpb,
       B, 256, 1, 512, 1024, B * 256);
    conv_layer(stream, in, wt, IN_FP3 + 0, fpb, gB, B * 256, 1536, 256);
    conv_layer(stream, in, wt, IN_FP3 + 4, gB, l2pn, B * 256, 256, 256);

    EW(interp_concat_kernel, B * 1024, l1x, l2x, l1p, l2pn, fpb, B, 1024, 256, 320, 256, B * 1024);
    conv_layer(stream, in, wt, IN_FP2 + 0, fpb, gB, B * 1024, 576, 256);
    conv_layer(stream, in, wt, IN_FP2 + 4, gB, l1pn, B * 1024, 256, 128);

    EW(build_p1_kernel, B * N0 * 22, in[IN_CLS], xb, p1b, N0, B * N0 * 22);
    EW(interp_concat_kernel, B * N0, xb, l1x, p1b, l1pn, fpb, B, N0, 1024, 22, 128, B * N0);
    conv_layer(stream, in, wt, IN_FP1 + 0, fpb, gB, B * N0, 150, 128);
    conv_layer(stream, in, wt, IN_FP1 + 4, gB, l0p, B * N0, 128, 128);

    // ---- conv1 -> feat; pooled -> count head ----
    conv_layer(stream, in, wt, IN_CONV1, l0p, feat, B * N0, 128, 128);
    EW(meanpool_kernel, B * 128, feat, pooled, B, N0, 128, B * 128);
    gemm(stream, pooled, wt[IN_COUNT_W1], in[IN_COUNT_B1], nullptr, nullptr, cnth,
         B, 64, 128, 128, 128, 64, 1.f, 1);
    gemm(stream, cnth, wt[IN_COUNT_W2], in[IN_COUNT_B2], nullptr, nullptr, fout + 16000,
         B, 1, 64, 64, 64, 1, 1.f, 0);

    // ---- transformer on queries (memory branch is dead code in the reference) ----
    const int Lq = 1000, Mq = Lq * B;
    EW(bc_query_kernel, Mq * 128, in[IN_QUERY], xq, B, Mq * 128);
    const float inv_sqrt_d = 0.08838834764831845f;  // 1/sqrt(128)
    for (int l = 0; l < 2; ++l) {
        const int t0 = IN_TR + l * 16;
        gemm(stream, xq, wt[t0 + 4], in[t0 + 10], nullptr, nullptr, qb, Mq, 128, 128, 128, 128, 128, 1.f, 0);
        gemm(stream, xq, wt[t0 + 2], in[t0 + 8],  nullptr, nullptr, kb, Mq, 128, 128, 128, 128, 128, 1.f, 0);
        // Vt[b][d][m] = sum_k Wv^T[d,k] * x[m,b,k] + bv[d]  (bias along rows)
        gemm(stream, wt[t0 + 5], xq, in[t0 + 11], nullptr, nullptr, vt,
             128, Lq, 128, 128, B * 128, Lq, 1.f, 0, /*biasM=*/1,
             B, 0, 128, (long long)128 * Lq);
        // scores[b,l,m] = q[l,b]·k[m,b] / sqrt(d)
        gemm(stream, qb, kb, nullptr, nullptr, nullptr, sc, Lq, Lq, 128,
             B * 128, B * 128, Lq, inv_sqrt_d, 0, 0, B, 128, 128, (long long)Lq * Lq);
        softmax_kernel<<<B * Lq, 256, 0, stream>>>(sc, Lq);
        // o[l,b,d] = sum_m attn[b,l,m] * Vt[b][d][m]
        gemm(stream, sc, vt, nullptr, nullptr, nullptr, ob, Lq, 128, Lq,
             Lq, Lq, B * 128, 1.f, 0, 0, B, (long long)Lq * Lq, (long long)128 * Lq, 128);
        gemm(stream, ob, wt[t0 + 3], in[t0 + 9], nullptr, nullptr, tb, Mq, 128, 128, 128, 128, 128, 1.f, 0);
        ln_res_kernel<<<Mq, 128, 0, stream>>>(xq, tb, in[t0 + 13], in[t0 + 12]);
        gemm(stream, xq, wt[t0 + 0], in[t0 + 6], nullptr, nullptr, hb, Mq, 512, 128, 128, 128, 512, 1.f, 1);
        gemm(stream, hb, wt[t0 + 1], in[t0 + 7], nullptr, nullptr, tb, Mq, 128, 512, 512, 512, 128, 1.f, 0);
        ln_res_kernel<<<Mq, 128, 0, stream>>>(xq, tb, in[t0 + 15], in[t0 + 14]);
    }
    EW(lb_to_bl_kernel, Mq * 128, xq, dbt, Lq, B, Mq * 128);

    // ---- heads: coords [B,Q,3], exist [B,Q], count [B,1] ----
    gemm(stream, dbt, wt[IN_COORD_W1], in[IN_COORD_B1], nullptr, nullptr, hd,
         Mq, 128, 128, 128, 128, 128, 1.f, 1);
    gemm(stream, hd, wt[IN_COORD_W2], in[IN_COORD_B2], nullptr, nullptr, fout,
         Mq, 3, 128, 128, 128, 3, 1.f, 0);
    gemm(stream, dbt, wt[IN_EXIST_W], in[IN_EXIST_B], nullptr, nullptr, fout + 12000,
         Mq, 1, 128, 128, 128, 1, 1.f, 2);
}
